// MultisimilarityCriterion_63187558859132
// MI455X (gfx1250) — matrix operand order repaired and checked by reading, not compiled, not run on hardware
//
#include <hip/hip_runtime.h>
#include <hip/hip_bf16.h>

typedef __attribute__((ext_vector_type(16))) _Float16 v16h;
typedef __attribute__((ext_vector_type(8)))  _Float16 v8h;
typedef __attribute__((ext_vector_type(8)))  float    v8f;

#define N_ROWS 4096
#define K_DIM  128
#define ROW_P  136   // padded LDS row pitch in halves (272B = 68 dwords -> bank skew)
#define POS_W  2.0f
#define NEG_W  40.0f
#define MARGIN 0.1f
#define THRESH 0.5f

// ---- monotone float <-> uint keys for atomic min/max on floats ----
__device__ __forceinline__ unsigned f2key(float f) {
  unsigned u = __float_as_uint(f);
  return (u & 0x80000000u) ? ~u : (u | 0x80000000u);
}
__device__ __forceinline__ float key2f(unsigned k) {
  unsigned u = (k & 0x80000000u) ? (k & 0x7fffffffu) : ~k;
  return __uint_as_float(u);
}

// ---- WMMA wrapper: D = A(16x32 f16) * B(32x16 f16) + C (f32 acc) ----
__device__ __forceinline__ v8f wmma_f16(v16h a, v16h b, v8f c) {
  return __builtin_amdgcn_wmma_f32_16x16x32_f16(
      /*neg_a=*/false, a, /*neg_b=*/false, b,
      /*c_mod=*/(short)0, c, /*reuse_a=*/false, /*reuse_b=*/false);
}

// ---- CDNA5 async global->LDS copy (16B per lane), tracked by ASYNCcnt ----
__device__ __forceinline__ void async_copy_b128(unsigned lds_off,
                                                const _Float16* gsrc) {
  asm volatile("global_load_async_to_lds_b128 %0, %1, off"
               :: "v"(lds_off), "v"(gsrc)
               : "memory");
}
__device__ __forceinline__ void wait_asynccnt0() {
  asm volatile("s_wait_asynccnt 0x0" ::: "memory");
}
// Flat LDS aperture: low 32 bits of a generic pointer to __shared__ are the
// LDS byte address (ISA 10.2).
__device__ __forceinline__ unsigned lds_addr(const void* p) {
  return (unsigned)(uintptr_t)p;
}

// A-fragments (hi+lo) for one 16-column tile, all 4 K-steps. 64 VGPRs.
struct AFrag {
  v16h ah[4];
  v16h al[4];
};

// Load A fragments from the padded LDS stage buffer (row = local column idx).
__device__ __forceinline__ void load_afrag_lds(AFrag& f,
                                               const _Float16* sh,
                                               const _Float16* sl,
                                               int row, int c0) {
  const _Float16* ph = sh + row * ROW_P;
  const _Float16* pl = sl + row * ROW_P;
#pragma unroll
  for (int ks = 0; ks < 4; ks++) {
    // per-lane K chunks: {c0..c0+7} and {c0+8..c0+15} (2x ds_load_b128)
    *(v8h*)&f.ah[ks]       = *(const v8h*)(ph + ks * 32 + c0);
    *((v8h*)&f.ah[ks] + 1) = *(const v8h*)(ph + ks * 32 + c0 + 8);
    *(v8h*)&f.al[ks]       = *(const v8h*)(pl + ks * 32 + c0);
    *((v8h*)&f.al[ks] + 1) = *(const v8h*)(pl + ks * 32 + c0 + 8);
  }
}

// Split-f16 tile: S ~= hi*hi + hi*lo + lo*hi (fp32 accumulate) -> 12 WMMAs.
__device__ __forceinline__ v8f compute_tile(const AFrag& f,
                                            const v16h* bhi, const v16h* blo) {
  v8f c = {};
#pragma unroll
  for (int ks = 0; ks < 4; ks++) {
    c = wmma_f16(f.ah[ks], bhi[ks], c);
    c = wmma_f16(f.ah[ks], blo[ks], c);
    c = wmma_f16(f.al[ks], bhi[ks], c);
  }
  return c;
}

// ---- Kernel 1: L2-normalize rows, write split-f16 (hi + lo) copies ----
__global__ void __launch_bounds__(32) k_normalize(const float* __restrict__ x,
                                                  _Float16* __restrict__ xhi,
                                                  _Float16* __restrict__ xlo) {
  const int row = blockIdx.x;
  const int lane = threadIdx.x;  // 32 lanes, 4 elements each
  const float* p = x + (size_t)row * K_DIM;
  float v[4];
  float ss = 0.f;
#pragma unroll
  for (int t = 0; t < 4; t++) { v[t] = p[lane + 32 * t]; ss += v[t] * v[t]; }
#pragma unroll
  for (int off = 16; off > 0; off >>= 1) ss += __shfl_xor(ss, off, 32);
  const float inv = 1.0f / sqrtf(ss);
#pragma unroll
  for (int t = 0; t < 4; t++) {
    const float xn = v[t] * inv;
    const _Float16 h = (_Float16)xn;
    xhi[(size_t)row * K_DIM + lane + 32 * t] = h;
    xlo[(size_t)row * K_DIM + lane + 32 * t] = (_Float16)(xn - (float)h);
  }
}

// ---- Kernel 2: init per-row reduction state ----
__global__ void k_init(unsigned* minkey, unsigned* maxkey,
                       float* psum, float* nsum, unsigned* flags) {
  const int i = blockIdx.x * blockDim.x + threadIdx.x;
  if (i < N_ROWS) {
    minkey[i] = f2key(1e9f);    // BIG sentinel, matches reference
    maxkey[i] = f2key(-1e9f);
    psum[i] = 0.f;
    nsum[i] = 0.f;
    flags[i] = 0u;
  }
}

// ---- Kernel 3 (pass 1): per-row hardest positive (min) / negative (max) ----
__global__ void __launch_bounds__(256) k_pass1(const _Float16* __restrict__ xhi,
                                               const _Float16* __restrict__ xlo,
                                               const int* __restrict__ labels,
                                               unsigned* __restrict__ minkey,
                                               unsigned* __restrict__ maxkey) {
  __shared__ int slab[N_ROWS];                               // 16 KB
  __shared__ __align__(16) _Float16 sbuf[2][2][32 * ROW_P];  // 2 buf x {hi,lo}, 34 KB
  for (int t = threadIdx.x; t < N_ROWS; t += blockDim.x) slab[t] = labels[t];

  const int lane = threadIdx.x & 31;
  const int wave = threadIdx.x >> 5;
  const int i0   = (blockIdx.x * 8 + wave) * 16;   // 16 loss-rows per wave (N dim)
  const int irow = i0 + (lane & 15);
  const int kb   = (lane < 16) ? 0 : 16;  // B fragment: contiguous K per lane half
  const int c0   = (lane < 16) ? 0 : 16;  // A fragment K chunk base
  const int jsub = (lane < 16) ? 0 : 8;   // D: lanes>=16 hold M=8..15
  const int m    = lane & 15;

  // Preload B fragments (fixed loss-row block) for all 4 K-steps, hi and lo.
  v16h bhi[4], blo[4];
  const _Float16* pbh = xhi + (size_t)irow * K_DIM + kb;
  const _Float16* pbl = xlo + (size_t)irow * K_DIM + kb;
#pragma unroll
  for (int ks = 0; ks < 4; ks++) {
    bhi[ks] = *(const v16h*)(pbh + ks * 32);
    blo[ks] = *(const v16h*)(pbl + ks * 32);
  }

  // Issue async stage of 32 columns (rows of x) into LDS buffer b.
  auto stage = [&](int b, int J) {
    const int tid = threadIdx.x;
#pragma unroll
    for (int q = 0; q < 2; ++q) {
      const int seg = tid + q * 256;  // 0..511: (row, 16B-chunk) of the 32x256B block
      const int row = seg >> 4;
      const int c16 = seg & 15;
      async_copy_b128(lds_addr(&sbuf[b][0][row * ROW_P + c16 * 8]),
                      xhi + (size_t)(J + row) * K_DIM + c16 * 8);
      async_copy_b128(lds_addr(&sbuf[b][1][row * ROW_P + c16 * 8]),
                      xlo + (size_t)(J + row) * K_DIM + c16 * 8);
    }
  };

  const int jc0 = blockIdx.y * 512;  // 8 column chunks of 512 = 16 iter x 32 cols
  int cur = 0;
  stage(0, jc0);
  wait_asynccnt0();
  __syncthreads();

  // Label table is complete after the first barrier: hoist own label to a reg.
  const int mylab = slab[irow];

  float minpos = 1e9f, maxneg = -1e9f;
  // Branchless reductions: select candidate vs. sentinel, then min/max.
  auto reduce_tile = [&](const v8f& c, int j0) {
    const int4 l0 = *(const int4*)&slab[j0 + jsub];
    const int4 l1 = *(const int4*)&slab[j0 + jsub + 4];
    const int labv[8] = {l0.x, l0.y, l0.z, l0.w, l1.x, l1.y, l1.z, l1.w};
#pragma unroll
    for (int r = 0; r < 8; r++) {
      const int j = j0 + jsub + r;
      const float s = c[r];
      const bool same = (labv[r] == mylab);
      const float posv = (same && j != irow) ? s : 1e9f;
      const float negv = same ? -1e9f : s;
      minpos = fminf(minpos, posv);
      maxneg = fmaxf(maxneg, negv);
    }
  };

  for (int it = 0; it < 16; ++it) {
    const int j0 = jc0 + it * 32;
    if (it + 1 < 16) stage(cur ^ 1, j0 + 32);  // uniform branch: EXEC stays full
    AFrag fA, fB;
    load_afrag_lds(fA, &sbuf[cur][0][0], &sbuf[cur][1][0], m, c0);
    load_afrag_lds(fB, &sbuf[cur][0][0], &sbuf[cur][1][0], 16 + m, c0);
    v8f cA = compute_tile(fA, bhi, blo);
    reduce_tile(cA, j0);
    v8f cB = compute_tile(fB, bhi, blo);
    reduce_tile(cB, j0 + 16);
    wait_asynccnt0();
    __syncthreads();
    cur ^= 1;
  }

  // lane L and L^16 process the same loss-row: combine, then one atomic.
  minpos = fminf(minpos, __shfl_xor(minpos, 16, 32));
  maxneg = fmaxf(maxneg, __shfl_xor(maxneg, 16, 32));
  if (lane < 16) {
    atomicMin(&minkey[irow], f2key(minpos));
    atomicMax(&maxkey[irow], f2key(maxneg));
  }
}

// ---- Kernel 4 (pass 2): mined exp-sums + validity flags ----
__global__ void __launch_bounds__(256) k_pass2(const _Float16* __restrict__ xhi,
                                               const _Float16* __restrict__ xlo,
                                               const int* __restrict__ labels,
                                               const unsigned* __restrict__ minkey,
                                               const unsigned* __restrict__ maxkey,
                                               float* __restrict__ psum,
                                               float* __restrict__ nsum,
                                               unsigned* __restrict__ flags) {
  __shared__ int slab[N_ROWS];
  __shared__ __align__(16) _Float16 sbuf[2][2][32 * ROW_P];
  for (int t = threadIdx.x; t < N_ROWS; t += blockDim.x) slab[t] = labels[t];

  const int lane = threadIdx.x & 31;
  const int wave = threadIdx.x >> 5;
  const int i0   = (blockIdx.x * 8 + wave) * 16;
  const int irow = i0 + (lane & 15);
  const int kb   = (lane < 16) ? 0 : 16;
  const int c0   = (lane < 16) ? 0 : 16;
  const int jsub = (lane < 16) ? 0 : 8;
  const int m    = lane & 15;

  v16h bhi[4], blo[4];
  const _Float16* pbh = xhi + (size_t)irow * K_DIM + kb;
  const _Float16* pbl = xlo + (size_t)irow * K_DIM + kb;
#pragma unroll
  for (int ks = 0; ks < 4; ks++) {
    bhi[ks] = *(const v16h*)(pbh + ks * 32);
    blo[ks] = *(const v16h*)(pbl + ks * 32);
  }

  auto stage = [&](int b, int J) {
    const int tid = threadIdx.x;
#pragma unroll
    for (int q = 0; q < 2; ++q) {
      const int seg = tid + q * 256;
      const int row = seg >> 4;
      const int c16 = seg & 15;
      async_copy_b128(lds_addr(&sbuf[b][0][row * ROW_P + c16 * 8]),
                      xhi + (size_t)(J + row) * K_DIM + c16 * 8);
      async_copy_b128(lds_addr(&sbuf[b][1][row * ROW_P + c16 * 8]),
                      xlo + (size_t)(J + row) * K_DIM + c16 * 8);
    }
  };

  const float minpos_i = key2f(minkey[irow]);
  const float maxneg_i = key2f(maxkey[irow]);

  const int jc0 = blockIdx.y * 512;
  int cur = 0;
  stage(0, jc0);
  wait_asynccnt0();
  __syncthreads();

  const int mylab = slab[irow];

  float ps = 0.f, ns = 0.f;
  unsigned fl = 0u;
  // Branchless: exp args are bounded (S in [-1,1]) so compute unconditionally
  // (co-executes with XDL WMMAs) and select the addend / flag bit.
  auto reduce_tile = [&](const v8f& c, int j0) {
    const int4 l0 = *(const int4*)&slab[j0 + jsub];
    const int4 l1 = *(const int4*)&slab[j0 + jsub + 4];
    const int labv[8] = {l0.x, l0.y, l0.z, l0.w, l1.x, l1.y, l1.z, l1.w};
#pragma unroll
    for (int r = 0; r < 8; r++) {
      const int j = j0 + jsub + r;
      const float s = c[r];
      const bool same = (labv[r] == mylab);
      const bool nsel = !same && (s + MARGIN > minpos_i);
      const bool psel = same && (j != irow) && (s - MARGIN < maxneg_i);
      const float en = expf(NEG_W * (s - THRESH));
      const float ep = expf(-POS_W * (s - THRESH));
      ns += nsel ? en : 0.f;
      ps += psel ? ep : 0.f;
      fl |= nsel ? 1u : 0u;
      fl |= psel ? 2u : 0u;
    }
  };

  for (int it = 0; it < 16; ++it) {
    const int j0 = jc0 + it * 32;
    if (it + 1 < 16) stage(cur ^ 1, j0 + 32);
    AFrag fA, fB;
    load_afrag_lds(fA, &sbuf[cur][0][0], &sbuf[cur][1][0], m, c0);
    load_afrag_lds(fB, &sbuf[cur][0][0], &sbuf[cur][1][0], 16 + m, c0);
    v8f cA = compute_tile(fA, bhi, blo);
    reduce_tile(cA, j0);
    v8f cB = compute_tile(fB, bhi, blo);
    reduce_tile(cB, j0 + 16);
    wait_asynccnt0();
    __syncthreads();
    cur ^= 1;
  }

  ps += __shfl_xor(ps, 16, 32);
  ns += __shfl_xor(ns, 16, 32);
  fl |= (unsigned)__shfl_xor((int)fl, 16, 32);
  if (lane < 16) {
    atomicAdd(&psum[irow], ps);
    atomicAdd(&nsum[irow], ns);
    atomicOr(&flags[irow], fl);
  }
}

// ---- Kernel 5: final scalar reduction (loss, pos_mean, neg_mean) ----
__global__ void __launch_bounds__(256) k_final(const float* __restrict__ psum,
                                               const float* __restrict__ nsum,
                                               const unsigned* __restrict__ flags,
                                               float* __restrict__ out) {
  __shared__ float sp[256];
  __shared__ float sn[256];
  __shared__ int   sc[256];
  float p = 0.f, n = 0.f;
  int cnt = 0;
  for (int i = threadIdx.x; i < N_ROWS; i += 256) {
    // valid == (mined-negative exists) && (mined-positive exists); these imply
    // pos_m.any/neg_m.any since selections are subsets of the raw masks.
    if ((flags[i] & 3u) == 3u) {
      p += log1pf(psum[i]) * (1.0f / POS_W);
      n += log1pf(nsum[i]) * (1.0f / NEG_W);
      cnt++;
    }
  }
  sp[threadIdx.x] = p; sn[threadIdx.x] = n; sc[threadIdx.x] = cnt;
  __syncthreads();
  for (int off = 128; off > 0; off >>= 1) {
    if (threadIdx.x < off) {
      sp[threadIdx.x] += sp[threadIdx.x + off];
      sn[threadIdx.x] += sn[threadIdx.x + off];
      sc[threadIdx.x] += sc[threadIdx.x + off];
    }
    __syncthreads();
  }
  if (threadIdx.x == 0) {
    const int cc = sc[0] > 0 ? sc[0] : 1;
    const float pm = sp[0] / (float)cc;
    const float nm = sn[0] / (float)cc;
    out[0] = pm + nm;  // loss
    out[1] = pm;       // pos_mean
    out[2] = nm;       // neg_mean
  }
}

extern "C" void kernel_launch(void* const* d_in, const int* in_sizes, int n_in,
                              void* d_out, int out_size, void* d_ws, size_t ws_size,
                              hipStream_t stream) {
  (void)in_sizes; (void)n_in; (void)out_size; (void)ws_size;
  const float* batch  = (const float*)d_in[0];
  const int*   labels = (const int*)d_in[1];
  float* out = (float*)d_out;

  // Workspace layout (~2.1 MB): split-f16 normalized x + per-row reduction state.
  char* w = (char*)d_ws;
  _Float16* xhi = (_Float16*)w;                                   // 1 MB
  _Float16* xlo = (_Float16*)(w + (size_t)N_ROWS * K_DIM * 2);    // 1 MB
  char* w2 = w + (size_t)N_ROWS * K_DIM * 4;
  unsigned* minkey = (unsigned*)(w2);
  unsigned* maxkey = (unsigned*)(w2 + N_ROWS * 4);
  float*    psum   = (float*)(w2 + N_ROWS * 8);
  float*    nsum   = (float*)(w2 + N_ROWS * 12);
  unsigned* flags  = (unsigned*)(w2 + N_ROWS * 16);

  k_normalize<<<N_ROWS, 32, 0, stream>>>(batch, xhi, xlo);
  k_init<<<N_ROWS / 256, 256, 0, stream>>>(minkey, maxkey, psum, nsum, flags);

  dim3 grid(N_ROWS / (16 * 8), 8);  // 32 row-groups (8 waves each) x 8 column chunks
  k_pass1<<<grid, 256, 0, stream>>>(xhi, xlo, labels, minkey, maxkey);
  k_pass2<<<grid, 256, 0, stream>>>(xhi, xlo, labels, minkey, maxkey, psum, nsum, flags);
  k_final<<<1, 256, 0, stream>>>(psum, nsum, flags, out);
}
